// GCN_10161892623037
// MI455X (gfx1250) — matrix operand order repaired and checked by reading, not compiled, not run on hardware
//
#include <hip/hip_runtime.h>

typedef __attribute__((ext_vector_type(16))) __bf16 v16bf;
typedef __attribute__((ext_vector_type(8)))  float  v8f;

#define D_HID 128

// ---------- elementwise: split f32 -> bf16 hi + bf16 lo (layer-0 input only) ----------
__global__ void cvt_split_kernel(const float* __restrict__ x,
                                 __bf16* __restrict__ hi,
                                 __bf16* __restrict__ lo, int n) {
  int i = blockIdx.x * blockDim.x + threadIdx.x;
  if (i < n) {
    float v = x[i];
    __bf16 h = (__bf16)v;
    hi[i] = h;
    lo[i] = (__bf16)(v - (float)h);
  }
}

// ---------- weight prep: W[K,N] f32 -> WT[N,K] bf16 hi/lo ----------
__global__ void cvt_weightT_kernel(const float* __restrict__ w,
                                   __bf16* __restrict__ hiT,
                                   __bf16* __restrict__ loT, int K, int N) {
  int i = blockIdx.x * blockDim.x + threadIdx.x;
  if (i < K * N) {
    int k = i / N, n = i % N;
    float v = w[i];
    __bf16 h = (__bf16)v;
    hiT[(size_t)n * K + k] = h;
    loT[(size_t)n * K + k] = (__bf16)(v - (float)h);
  }
}

// ---------- GEMM: C[M,N] = A[M,128] * B[128,N] (+bias), bf16x3 split WMMA ----------
// A given as bf16 hi/lo row-major [M,128]; B given transposed bf16 hi/lo [N,128].
// One wave per 16x16 tile; wave w of the block covers column-tile w; block x = row tile.
__global__ void gemm_bf16x3_kernel(const __bf16* __restrict__ Ahi, const __bf16* __restrict__ Alo,
                                   const __bf16* __restrict__ BThi, const __bf16* __restrict__ BTlo,
                                   const float* __restrict__ bias,
                                   float* __restrict__ C, float* __restrict__ C2,
                                   int M, int N) {
  const int K = D_HID;
  const int wave  = threadIdx.x >> 5;
  const int lane  = threadIdx.x & 31;
  const int tileM = blockIdx.x;
  const int tileN = wave;
  const int half  = lane >> 4;      // lane-half select
  const int r     = lane & 15;

  const __bf16* arow_h = Ahi  + (size_t)(tileM * 16 + r) * K;
  const __bf16* arow_l = Alo  + (size_t)(tileM * 16 + r) * K;
  const __bf16* bcol_h = BThi + (size_t)(tileN * 16 + r) * K;
  const __bf16* bcol_l = BTlo + (size_t)(tileN * 16 + r) * K;

  v8f acc = {};
#pragma unroll
  for (int kc = 0; kc < K / 32; ++kc) {
    const int kb = kc * 32;
    v16bf ah, al, bh, bl;
#pragma unroll
    for (int v = 0; v < 8; ++v) {
      // A 16x32 bf16 layout: VGPR v -> K = (v>>2)*16 + (v&3)*2 + half*8 + {0,1}
      const int ka = kb + ((v >> 2) << 4) + ((v & 3) << 1) + (half << 3);
      ah[2 * v]     = arow_h[ka];
      ah[2 * v + 1] = arow_h[ka + 1];
      al[2 * v]     = arow_l[ka];
      al[2 * v + 1] = arow_l[ka + 1];
      // B 32x16 bf16 layout: VGPR v -> K = (v>>2)*16 + (v&3)*2 + half*8 + {0,1}
      const int kkb = kb + ((v >> 2) << 4) + ((v & 3) << 1) + (half << 3);
      bh[2 * v]     = bcol_h[kkb];
      bh[2 * v + 1] = bcol_h[kkb + 1];
      bl[2 * v]     = bcol_l[kkb];
      bl[2 * v + 1] = bcol_l[kkb + 1];
    }
    acc = __builtin_amdgcn_wmma_f32_16x16x32_bf16(false, ah, false, bh, (short)0, acc, false, false);
    acc = __builtin_amdgcn_wmma_f32_16x16x32_bf16(false, al, false, bh, (short)0, acc, false, false);
    acc = __builtin_amdgcn_wmma_f32_16x16x32_bf16(false, ah, false, bl, (short)0, acc, false, false);
  }

  const int col = tileN * 16 + r;
  const float b = bias ? bias[col] : 0.0f;
#pragma unroll
  for (int rr = 0; rr < 8; ++rr) {
    const int m = tileM * 16 + rr + half * 8;   // C/D: VGPR rr -> M=rr (+8 for upper lanes)
    const float v = acc[rr] + b;
    C[(size_t)m * N + col] = v;
    if (C2) C2[(size_t)m * N + col] = v;        // self-loop contribution into aggr
  }
}

// ---------- degree counts ----------
__global__ void count_init_kernel(int* __restrict__ cnt, int n) {
  int i = blockIdx.x * blockDim.x + threadIdx.x;
  if (i < n) cnt[i] = 1;                         // self-loop
}
__global__ void count_edges_kernel(const int* __restrict__ dst, int* __restrict__ cnt, int E) {
  int i = blockIdx.x * blockDim.x + threadIdx.x;
  if (i < E) atomicAdd(&cnt[dst[i]], 1);
}
__global__ void inv_count_kernel(const int* __restrict__ cnt, float* __restrict__ inv, int n) {
  int i = blockIdx.x * blockDim.x + threadIdx.x;
  if (i < n) inv[i] = 1.0f / (float)cnt[i];
}

// ---------- edge scatter: one wave per edge; lane owns 4 contiguous cols (b128 load) ----------
__global__ void scatter_kernel(const float4* __restrict__ h4,
                               const int* __restrict__ src, const int* __restrict__ dst,
                               float* __restrict__ aggr, int E) {
  int gid  = blockIdx.x * blockDim.x + threadIdx.x;
  int e    = gid >> 5;
  int lane = gid & 31;
  if (e >= E) return;
  int s = src[e], d = dst[e];
  float4 v = h4[(size_t)s * 32 + lane];          // one global_load_b128 per lane
  float* a = aggr + (size_t)d * D_HID + lane * 4;
  atomicAdd(a + 0, v.x);
  atomicAdd(a + 1, v.y);
  atomicAdd(a + 2, v.z);
  atomicAdd(a + 3, v.w);
}

// ---------- fused: x_next = relu(aggr*inv + lin), emitted directly as bf16 hi/lo ----------
__global__ void finalize_split_kernel(const float* __restrict__ aggr, const float* __restrict__ lin,
                                      const float* __restrict__ inv,
                                      __bf16* __restrict__ xhi, __bf16* __restrict__ xlo, int n) {
  int i = blockIdx.x * blockDim.x + threadIdx.x;
  if (i < n) {
    int row = i >> 7;                            // /128
    float v = fmaf(aggr[i], inv[row], lin[i]);
    v = v > 0.0f ? v : 0.0f;
    __bf16 h = (__bf16)v;
    xhi[i] = h;
    xlo[i] = (__bf16)(v - (float)h);
  }
}

// ---------- log_softmax over 64 cols, one wave per row ----------
__global__ void logsoftmax_kernel(const float* __restrict__ logits, float* __restrict__ out, int Nn) {
  int gid  = blockIdx.x * blockDim.x + threadIdx.x;
  int row  = gid >> 5;
  int lane = gid & 31;
  if (row >= Nn) return;
  const float* l = logits + (size_t)row * 64;
  float v0 = l[lane], v1 = l[lane + 32];
  float m = fmaxf(v0, v1);
#pragma unroll
  for (int o = 16; o > 0; o >>= 1) m = fmaxf(m, __shfl_xor(m, o, 32));
  float s = expf(v0 - m) + expf(v1 - m);
#pragma unroll
  for (int o = 16; o > 0; o >>= 1) s += __shfl_xor(s, o, 32);
  float ls = logf(s);
  out[(size_t)row * 64 + lane]      = v0 - m - ls;
  out[(size_t)row * 64 + lane + 32] = v1 - m - ls;
}

extern "C" void kernel_launch(void* const* d_in, const int* in_sizes, int n_in,
                              void* d_out, int out_size, void* d_ws, size_t ws_size,
                              hipStream_t stream) {
  (void)n_in; (void)out_size; (void)ws_size;

  const float* x0   = (const float*)d_in[0];
  const int*   eidx = (const int*)d_in[1];
  const float* w1   = (const float*)d_in[2];
  const float* l1w  = (const float*)d_in[3];
  const float* l1b  = (const float*)d_in[4];
  const float* w2   = (const float*)d_in[5];
  const float* l2w  = (const float*)d_in[6];
  const float* l2b  = (const float*)d_in[7];
  const float* low  = (const float*)d_in[8];
  const float* lob  = (const float*)d_in[9];

  const int Nn   = in_sizes[0] / D_HID;   // 50000
  const int E    = in_sizes[1] / 2;       // 600000
  const int Dout = in_sizes[9];           // 64
  const int* src = eidx;
  const int* dst = eidx + E;

  // ---- workspace carve-up ----
  char* ws = (char*)d_ws;
  size_t off = 0;
  auto take = [&](size_t bytes) -> char* {
    char* p = ws + off;
    off = (off + bytes + 255) & ~(size_t)255;
    return p;
  };
  const size_t nd = (size_t)Nn * D_HID;
  __bf16* xhi  = (__bf16*)take(nd * 2);
  __bf16* xlo  = (__bf16*)take(nd * 2);
  float*  hbuf = (float*)take(nd * 4);    // also reused for final logits (Nn x 64)
  float*  lin  = (float*)take(nd * 4);
  float*  aggr = (float*)take(nd * 4);
  int*    cnt  = (int*)take((size_t)Nn * 4);
  float*  inv  = (float*)take((size_t)Nn * 4);
  __bf16* w1Th = (__bf16*)take(128 * 128 * 2);
  __bf16* w1Tl = (__bf16*)take(128 * 128 * 2);
  __bf16* l1Th = (__bf16*)take(128 * 128 * 2);
  __bf16* l1Tl = (__bf16*)take(128 * 128 * 2);
  __bf16* w2Th = (__bf16*)take(128 * 128 * 2);
  __bf16* w2Tl = (__bf16*)take(128 * 128 * 2);
  __bf16* l2Th = (__bf16*)take(128 * 128 * 2);
  __bf16* l2Tl = (__bf16*)take(128 * 128 * 2);
  __bf16* loTh = (__bf16*)take(128 * 64 * 2);
  __bf16* loTl = (__bf16*)take(128 * 64 * 2);

  const int TB = 256;
  const int nElem   = Nn * D_HID;
  const int gElem   = (nElem + TB - 1) / TB;
  const int gNodes  = (Nn + TB - 1) / TB;
  const int gEdges  = (E + TB - 1) / TB;
  const int gEWaves = ((E * 32) + TB - 1) / TB;
  const int gRows   = ((Nn * 32) + TB - 1) / TB;
  const int gTilesM = Nn / 16;            // 3125, exact

  // ---- weight prep ----
  const int gW = (128 * 128 + TB - 1) / TB;
  cvt_weightT_kernel<<<gW, TB, 0, stream>>>(w1,  w1Th, w1Tl, 128, 128);
  cvt_weightT_kernel<<<gW, TB, 0, stream>>>(l1w, l1Th, l1Tl, 128, 128);
  cvt_weightT_kernel<<<gW, TB, 0, stream>>>(w2,  w2Th, w2Tl, 128, 128);
  cvt_weightT_kernel<<<gW, TB, 0, stream>>>(l2w, l2Th, l2Tl, 128, 128);
  cvt_weightT_kernel<<<(128 * 64 + TB - 1) / TB, TB, 0, stream>>>(low, loTh, loTl, 128, Dout);

  // ---- degrees (shared by both layers) ----
  count_init_kernel<<<gNodes, TB, 0, stream>>>(cnt, Nn);
  count_edges_kernel<<<gEdges, TB, 0, stream>>>(dst, cnt, E);
  inv_count_kernel<<<gNodes, TB, 0, stream>>>(cnt, inv, Nn);

  // ---- layer 1 ----
  cvt_split_kernel<<<gElem, TB, 0, stream>>>(x0, xhi, xlo, nElem);
  gemm_bf16x3_kernel<<<gTilesM, 8 * 32, 0, stream>>>(xhi, xlo, w1Th, w1Tl, nullptr, hbuf, aggr, Nn, D_HID);
  gemm_bf16x3_kernel<<<gTilesM, 8 * 32, 0, stream>>>(xhi, xlo, l1Th, l1Tl, l1b, lin, nullptr, Nn, D_HID);
  scatter_kernel<<<gEWaves, TB, 0, stream>>>((const float4*)hbuf, src, dst, aggr, E);
  finalize_split_kernel<<<gElem, TB, 0, stream>>>(aggr, lin, inv, xhi, xlo, nElem);

  // ---- layer 2 (xhi/xlo now hold x1) ----
  gemm_bf16x3_kernel<<<gTilesM, 8 * 32, 0, stream>>>(xhi, xlo, w2Th, w2Tl, nullptr, hbuf, aggr, Nn, D_HID);
  gemm_bf16x3_kernel<<<gTilesM, 8 * 32, 0, stream>>>(xhi, xlo, l2Th, l2Tl, l2b, lin, nullptr, Nn, D_HID);
  scatter_kernel<<<gEWaves, TB, 0, stream>>>((const float4*)hbuf, src, dst, aggr, E);
  finalize_split_kernel<<<gElem, TB, 0, stream>>>(aggr, lin, inv, xhi, xlo, nElem);

  // ---- output head: logits = x2 @ lin_out_w + b; log_softmax ----
  gemm_bf16x3_kernel<<<gTilesM, 4 * 32, 0, stream>>>(xhi, xlo, loTh, loTl, lob, hbuf, nullptr, Nn, Dout);
  logsoftmax_kernel<<<gRows, TB, 0, stream>>>(hbuf, (float*)d_out, Nn);
}
